// ResizeConv2D_88330297410401
// MI455X (gfx1250) — matrix operand order repaired and checked by reading, not compiled, not run on hardware
//
#include <hip/hip_runtime.h>
#include <math.h>

// ---------------------------------------------------------------------------
// StyleGAN2 upsample_conv_2d (2x transposed 3x3 conv, C=512) + 4-tap FIR blur
// MI455X / gfx1250: wave32, WMMA 16x16x32 bf16 (f32 accum), async LDS staging.
//
// Pipeline (all in d_ws):
//   zero_page : 1KB zeros (border reads resolve here -> no EXEC masking)
//   prep_w : w (3,3,512,512) f32 -> A[tap][co][ci] bf16, pre-scaled 1/sqrt(4608)
//   prep_x : x (8,512,64,64) f32 NCHW -> X [n][h][w][ci] bf16 (LDS transpose)
//   upconv : 4 polyphase GEMMs, double-buffered GLOBAL_LOAD_ASYNC_TO_LDS
//            staging + v_wmma -> Y [n][129][129][co] bf16
//   blur   : 4x4 binomial FIR (LDS halo tile) -> out f32 NCHW (8,512,128,128)
// ---------------------------------------------------------------------------

typedef __bf16 bf16;
typedef __attribute__((ext_vector_type(16))) __bf16 v16bf;
typedef __attribute__((ext_vector_type(8)))  __bf16 v8bf;
typedef __attribute__((ext_vector_type(8)))  float  v8f;
typedef __attribute__((ext_vector_type(4)))  int    v4i;

#define CIN  512
#define COUT 512
#define HY   129
#define WY   129
#define NBATCH 8

// workspace layout (bytes)
#define OFF_A 0
#define SZ_A  ((size_t)9 * 512 * 512 * 2)                 //   4.7 MB
#define OFF_X (OFF_A + SZ_A)
#define SZ_X  ((size_t)NBATCH * 64 * 64 * 512 * 2)        //  33.6 MB
#define OFF_Y (OFF_X + SZ_X)
#define SZ_Y  ((size_t)NBATCH * HY * WY * 512 * 2)        // 136.3 MB
#define OFF_Z (OFF_Y + SZ_Y)                              // 1 KB zero page

// ---- CDNA5 async global->LDS path (probe-guarded) -------------------------
#if defined(__HIP_DEVICE_COMPILE__) && \
    __has_builtin(__builtin_amdgcn_global_load_async_to_lds_b128)
#define HAVE_ASYNC 1
#else
#define HAVE_ASYNC 0
#endif

#define AS3P __attribute__((address_space(3))) void*
typedef __attribute__((address_space(3))) v4i* lds_v4i_p;

#if HAVE_ASYNC
#if __has_builtin(__builtin_amdgcn_s_wait_asynccnt)
#define WAIT_ASYNC(N) __builtin_amdgcn_s_wait_asynccnt(N)
#else
#define WAIT_ASYNC(N) asm volatile("s_wait_asynccnt " #N ::: "memory")
#endif
#else
#define WAIT_ASYNC(N)
#endif

// ---------------------------------------------------------------------------
__global__ __launch_bounds__(256) void zero_page_k(bf16* __restrict__ z) {
  int t = blockIdx.x * 256 + threadIdx.x;
  if (t < 512) z[t] = (bf16)0.0f;
}

// ---------------------------------------------------------------------------
// Weight prep: w[kh][kw][ci][co] f32 -> A[tap][co][ci] bf16 * runtime_coef
// ---------------------------------------------------------------------------
__global__ __launch_bounds__(256) void prep_w(const float* __restrict__ w,
                                              bf16* __restrict__ A, float coef) {
  int idx = blockIdx.x * 256 + threadIdx.x;   // exactly 9*512*512 threads
  int ci  = idx & 511;
  int co  = (idx >> 9) & 511;
  int tap = idx >> 18;
  if (tap >= 9) return;
  float v = w[((size_t)tap * 512 + ci) * 512 + co] * coef;
  A[((size_t)tap * 512 + co) * 512 + ci] = (bf16)v;
}

// ---------------------------------------------------------------------------
// Activation prep: x NCHW f32 -> X [n][h][w][ci] bf16 via LDS transpose.
// ---------------------------------------------------------------------------
__global__ __launch_bounds__(256) void prep_x(const float* __restrict__ x,
                                              bf16* __restrict__ X) {
  __shared__ __align__(16) bf16 tile[64 * 32];   // [w][ci]
  int h  = blockIdx.x;
  int cc = blockIdx.y;            // ci chunk (32 each)
  int n  = blockIdx.z;
  int t  = threadIdx.x;
  int wcol  = t & 63;
  int cbase = t >> 6;             // 0..3
  for (int it = 0; it < 8; ++it) {
    int ciOff = it * 4 + cbase;
    int ci = cc * 32 + ciOff;
    float v = x[(((size_t)n * 512 + ci) * 64 + h) * 64 + wcol];
    tile[wcol * 32 + ciOff] = (bf16)v;
  }
  __syncthreads();
  int w2  = t >> 2;
  int grp = t & 3;
  v8bf vv = *(const v8bf*)&tile[w2 * 32 + grp * 8];
  *(v8bf*)&X[((((size_t)n * 64 + h) * 64) + w2) * 512 + cc * 32 + grp * 8] = vv;
}

// ---------------------------------------------------------------------------
// Polyphase transposed-conv GEMM with async-LDS double buffering.
// Phase (ph,pw): y[2r+ph][2c+pw]; taps: ph==0 -> kh {0(xr=r-1), 2(xr=r)},
// ph==1 -> kh=1(xr=r); same per column.
// WG = 256 thr (8 wave32) computing 128co x 128px; wave tile 64co x 32px.
// Per 32-ci K-chunk: stage A(128x32) + B(128x32) bf16 tiles into LDS
// (rows padded to 40 elts = 80 B so the 16 lanes of a fragment read hit
// disjoint bank quads), then 8 v_wmma_f32_16x16x32_bf16.
// ---------------------------------------------------------------------------
__global__ __launch_bounds__(256) void upconv_phase(
    const bf16* __restrict__ A, const bf16* __restrict__ X,
    const bf16* __restrict__ Z, bf16* __restrict__ Y, int ph, int pw) {
  __shared__ __align__(16) bf16 shA[2][128 * 40];   // 2 x 10 KB
  __shared__ __align__(16) bf16 shB[2][128 * 40];   // 2 x 10 KB

  const int nRows = (ph == 0) ? 65 : 64;
  const int nCols = (pw == 0) ? 65 : 64;
  const int nPix  = nRows * nCols;

  const int n    = blockIdx.z;
  const int co0  = blockIdx.y * 128;
  const int pix0 = blockIdx.x * 128;

  const int tid   = threadIdx.x;
  const int lane  = tid & 31;
  const int wid   = tid >> 5;
  const int l16   = lane & 15;
  const int halfA = (lane >> 4) * 8;    // A-frag K base: 0 or 8   (doc 7.12.2)
  const int halfB = (lane >> 4) * 8;    // B-frag K base: 0 or 8
  const int halfD = (lane >> 4) * 8;    // D-frag M (co) base: 0 or 8

  const int waveCoL  = (wid & 1) * 64;
  const int wavePixL = (wid >> 1) * 32;

  // staging role: thread moves 4 x 16B per chunk (2 A rows + 2 B rows)
  const int sgrp = tid & 3;             // 16B group inside a 32-ci row
  const int srow = tid >> 2;            // 0..63  (rows srow, srow+64)
  const int aL0  = srow * 40 + sgrp * 8;
  const int aL1  = (srow + 64) * 40 + sgrp * 8;

  // taps for this phase
  int nh = (ph == 0) ? 2 : 1, nw = (pw == 0) ? 2 : 1;
  int khv[2], hof[2], kwv[2], wof[2];
  if (ph == 0) { khv[0] = 0; hof[0] = -1; khv[1] = 2; hof[1] = 0; }
  else         { khv[0] = 1; hof[0] = 0; }
  if (pw == 0) { kwv[0] = 0; wof[0] = -1; kwv[1] = 2; wof[1] = 0; }
  else         { kwv[0] = 1; wof[0] = 0; }

  // staging pixel rows (r,c) for this thread's two B rows
  int srP[2], scP[2]; bool sOK[2];
  for (int q = 0; q < 2; ++q) {
    int p = pix0 + q * 64 + srow;
    sOK[q] = (p < nPix);
    unsigned pp = sOK[q] ? (unsigned)p : 0u;
    srP[q] = (int)(pp / (unsigned)nCols);
    scP[q] = (int)(pp % (unsigned)nCols);
  }

  // compute-side pixel mapping (store phase)
  int pr[2], pc[2]; bool pok[2];
  for (int j = 0; j < 2; ++j) {
    int p = pix0 + wavePixL + j * 16 + l16;
    pok[j] = (p < nPix);
    unsigned pp = pok[j] ? (unsigned)p : 0u;
    pr[j] = (int)(pp / (unsigned)nCols);
    pc[j] = (int)(pp % (unsigned)nCols);
  }

  v8f acc[4][2];
  for (int i = 0; i < 4; ++i)
    for (int j = 0; j < 2; ++j) acc[i][j] = (v8f)(0.0f);

  int buf = 0;
  for (int th = 0; th < nh; ++th) {
    for (int tw = 0; tw < nw; ++tw) {
      const int tap = khv[th] * 3 + kwv[tw];
      // this thread's global staging pointers for this tap
      const bf16* aG = A + (size_t)tap * CIN * COUT
                         + (size_t)(co0 + srow) * CIN + sgrp * 8;
      const bf16* bG[2];
      for (int q = 0; q < 2; ++q) {
        int xr = srP[q] + hof[th];
        int xc = scP[q] + wof[tw];
        bool ok = sOK[q] && ((unsigned)xr < 64u) && ((unsigned)xc < 64u);
        bG[q] = ok ? (X + ((((size_t)n * 64 + xr) * 64) + xc) * CIN + sgrp * 8)
                   : (Z + sgrp * 8);   // zero page: real zeros, no EXEC games
      }

      auto stage = [&](int b, int ciOff) {
#if HAVE_ASYNC
        __builtin_amdgcn_global_load_async_to_lds_b128(
            (v4i*)(aG + ciOff),
            (lds_v4i_p)(AS3P)&shA[b][aL0], 0, 0);
        __builtin_amdgcn_global_load_async_to_lds_b128(
            (v4i*)(aG + 64 * CIN + ciOff),
            (lds_v4i_p)(AS3P)&shA[b][aL1], 0, 0);
        __builtin_amdgcn_global_load_async_to_lds_b128(
            (v4i*)(bG[0] + ciOff),
            (lds_v4i_p)(AS3P)&shB[b][aL0], 0, 0);
        __builtin_amdgcn_global_load_async_to_lds_b128(
            (v4i*)(bG[1] + ciOff),
            (lds_v4i_p)(AS3P)&shB[b][aL1], 0, 0);
#else
        *(v8bf*)&shA[b][aL0] = *(const v8bf*)(aG + ciOff);
        *(v8bf*)&shA[b][aL1] = *(const v8bf*)(aG + 64 * CIN + ciOff);
        *(v8bf*)&shB[b][aL0] = *(const v8bf*)(bG[0] + ciOff);
        *(v8bf*)&shB[b][aL1] = *(const v8bf*)(bG[1] + ciOff);
#endif
      };

      stage(buf, 0);                       // prologue for this tap
      for (int kc = 0; kc < 16; ++kc) {
        if (kc < 15) stage(buf ^ 1, (kc + 1) * 32);   // overlap next DMA
        if (kc < 15) { WAIT_ASYNC(4); } else { WAIT_ASYNC(0); }
        __syncthreads();                   // staged tiles visible WG-wide

        const bf16* Ab = &shA[buf][0];
        const bf16* Bb = &shB[buf][0];
        v16bf afrag[4];
#pragma unroll
        for (int i = 0; i < 4; ++i) {
          int base = (waveCoL + i * 16 + l16) * 40 + halfA;
          v8bf lo = *(const v8bf*)(Ab + base);
          v8bf hi = *(const v8bf*)(Ab + base + 16);
          v16bf a;
#pragma unroll
          for (int k = 0; k < 8; ++k) { a[k] = lo[k]; a[8 + k] = hi[k]; }
          afrag[i] = a;
        }
        v16bf bfrag[2];
#pragma unroll
        for (int j = 0; j < 2; ++j) {
          int base = (wavePixL + j * 16 + l16) * 40 + halfB;
          v8bf lo = *(const v8bf*)(Bb + base);
          v8bf hi = *(const v8bf*)(Bb + base + 16);
          v16bf b;
#pragma unroll
          for (int k = 0; k < 8; ++k) { b[k] = lo[k]; b[8 + k] = hi[k]; }
          bfrag[j] = b;
        }
#pragma unroll
        for (int i = 0; i < 4; ++i)
#pragma unroll
          for (int j = 0; j < 2; ++j)
            acc[i][j] = __builtin_amdgcn_wmma_f32_16x16x32_bf16(
                false, afrag[i], false, bfrag[j],
                (short)0, acc[i][j], false, false);

        __syncthreads();                   // buffer reuse fence
        buf ^= 1;
      }
    }
  }

  // store D: lane pixel = l16's pixel, 8 contiguous co per acc tile
  for (int j = 0; j < 2; ++j) {
    if (!pok[j]) continue;
    int oh = 2 * pr[j] + ph;
    int ow = 2 * pc[j] + pw;
    size_t base = ((((size_t)n * HY + oh) * WY) + ow) * (size_t)COUT;
    for (int i = 0; i < 4; ++i) {
      v8bf o;
#pragma unroll
      for (int k = 0; k < 8; ++k) o[k] = (bf16)acc[i][j][k];
      *(v8bf*)&Y[base + co0 + waveCoL + i * 16 + halfD] = o;
    }
  }
}

// ---------------------------------------------------------------------------
// 4x4 binomial FIR blur: out[oh][ow] = (1/16) * sum k[i]k[j]*y[oh-1+i][ow-1+j]
// (k = [1,3,3,1]; scale = GAIN*FACTOR^2/64 = 1/16). LDS halo tile 19x19x32ch.
// ---------------------------------------------------------------------------
__global__ __launch_bounds__(256) void blur_kernel(const bf16* __restrict__ Y,
                                                   float* __restrict__ out) {
  __shared__ __align__(16) bf16 sy[19 * 19 * 32];
  const int tileIdx = blockIdx.x;
  const int co0 = blockIdx.y * 32;
  const int n   = blockIdx.z;
  const int ty = tileIdx >> 3, tx = tileIdx & 7;
  const int oh0 = ty * 16, ow0 = tx * 16;
  const int t = threadIdx.x;

  for (int u = t; u < 19 * 19 * 4; u += 256) {
    int pos = u >> 2, sub = u & 3;
    int ry = pos / 19, rx = pos % 19;
    int yh = oh0 - 1 + ry;
    int yw = ow0 - 1 + rx;
    v8bf v = {};
    if ((unsigned)yh < (unsigned)HY && (unsigned)yw < (unsigned)WY)
      v = *(const v8bf*)&Y[((((size_t)n * HY + yh) * WY) + yw) * COUT
                           + co0 + sub * 8];
    *(v8bf*)&sy[pos * 32 + sub * 8] = v;
  }
  __syncthreads();

  const int lx = t & 15, ly = (t >> 4) & 15;
  const int ow = ow0 + lx, oh = oh0 + ly;
  const float wk[4] = {1.f, 3.f, 3.f, 1.f};

  float accv[32];
#pragma unroll
  for (int c = 0; c < 32; ++c) accv[c] = 0.f;
  for (int i = 0; i < 4; ++i) {
    for (int jj = 0; jj < 4; ++jj) {
      float wij = wk[i] * wk[jj];
      const bf16* p = &sy[(((ly + i) * 19) + (lx + jj)) * 32];
#pragma unroll
      for (int c = 0; c < 32; ++c) accv[c] += wij * (float)p[c];
    }
  }
  const float scale = 1.0f / 16.0f;
#pragma unroll
  for (int c = 0; c < 32; ++c)
    out[((((size_t)n * COUT + co0 + c) * 128) + oh) * 128 + ow] =
        accv[c] * scale;
}

// ---------------------------------------------------------------------------
extern "C" void kernel_launch(void* const* d_in, const int* in_sizes, int n_in,
                              void* d_out, int out_size, void* d_ws,
                              size_t ws_size, hipStream_t stream) {
  (void)in_sizes; (void)n_in; (void)out_size; (void)ws_size;
  const float* x = (const float*)d_in[0];
  const float* w = (const float*)d_in[1];
  float* out = (float*)d_out;
  char* ws = (char*)d_ws;
  bf16* A = (bf16*)(ws + OFF_A);
  bf16* X = (bf16*)(ws + OFF_X);
  bf16* Y = (bf16*)(ws + OFF_Y);
  bf16* Z = (bf16*)(ws + OFF_Z);

  const float coef = 1.0f / sqrtf(9.0f * 512.0f);   // runtime_coef

  zero_page_k<<<dim3(2), 256, 0, stream>>>(Z);
  prep_w<<<dim3((9 * 512 * 512) / 256), 256, 0, stream>>>(w, A, coef);
  prep_x<<<dim3(64, 16, 8), 256, 0, stream>>>(x, X);

  for (int ph = 0; ph < 2; ++ph)
    for (int pw = 0; pw < 2; ++pw) {
      int nRows = (ph == 0) ? 65 : 64;
      int nCols = (pw == 0) ? 65 : 64;
      int tiles = (nRows * nCols + 127) / 128;
      upconv_phase<<<dim3(tiles, COUT / 128, NBATCH), 256, 0, stream>>>(
          A, X, Z, Y, ph, pw);
    }

  blur_kernel<<<dim3(64, COUT / 32, NBATCH), 256, 0, stream>>>(Y, out);
}
